// RNNDecoder_73151882985808
// MI455X (gfx1250) — compile-verified, matrix operand order checked by reading, not run on hardware
//
#include <hip/hip_runtime.h>
#include <hip/hip_bf16.h>

// Problem dims
#define B_ 16384
#define H_ 512
#define F_ 2048
#define V_ 1000

typedef __attribute__((ext_vector_type(16))) __bf16 v16bf;
typedef __attribute__((ext_vector_type(8)))  float  v8f;
typedef __attribute__((ext_vector_type(4)))  int    v4i;

#if __has_builtin(__builtin_amdgcn_global_load_async_to_lds_b128) && \
    __has_builtin(__builtin_amdgcn_s_wait_asynccnt)
#define HAS_ASYNC 1
#else
#define HAS_ASYNC 0
#endif

__device__ __forceinline__ unsigned short f2bf(float f) {
  union { float f; unsigned int u; } v; v.f = f;
  unsigned int u = v.u;
  u += 0x7fffu + ((u >> 16) & 1u);   // round-to-nearest-even
  return (unsigned short)(u >> 16);
}

// 16-byte global -> LDS copy; async (ASYNCcnt) when available.
__device__ __forceinline__ void async_copy16(const unsigned short* g, unsigned short* l) {
#if HAS_ASYNC
  __builtin_amdgcn_global_load_async_to_lds_b128(
      (__attribute__((address_space(1))) v4i*)g,
      (__attribute__((address_space(3))) v4i*)l, 0, 0);
#else
  *(uint4*)l = *(const uint4*)g;
#endif
}

#if HAS_ASYNC
#define WAIT_ASYNC(n) __builtin_amdgcn_s_wait_asynccnt(n)
#else
#define WAIT_ASYNC(n) ((void)0)
#endif

// ---------------- prep kernels ----------------

__global__ void k_f2bf(const float* __restrict__ src, unsigned short* __restrict__ dst, size_t n) {
  size_t i = (size_t)blockIdx.x * blockDim.x + threadIdx.x;
  if (i < n) dst[i] = f2bf(src[i]);
}

// src [K,N] f32 row-major  ->  dst [N,K] bf16 row-major (i.e. W^T)
__global__ void k_transpose_bf(const float* __restrict__ src, unsigned short* __restrict__ dst,
                               int K, int N) {
  size_t i = (size_t)blockIdx.x * blockDim.x + threadIdx.x;
  size_t total = (size_t)K * N;
  if (i < total) {
    int n = (int)(i / K);
    int k = (int)(i % K);
    dst[i] = f2bf(src[(size_t)k * N + n]);
  }
}

__global__ void k_gather_bf(const int* __restrict__ lab, const float* __restrict__ emb,
                            unsigned short* __restrict__ wk) {
  size_t i = (size_t)blockIdx.x * blockDim.x + threadIdx.x;
  if (i < (size_t)B_ * H_) {
    int b = (int)(i >> 9);          // /512
    int h = (int)(i & (H_ - 1));
    wk[i] = f2bf(emb[(size_t)lab[b] * H_ + h]);
  }
}

__global__ void k_vadd(const float* __restrict__ a, const float* __restrict__ b,
                       float* __restrict__ dst, int n) {
  int i = blockIdx.x * blockDim.x + threadIdx.x;
  if (i < n) dst[i] = a[i] + b[i];
}

// cur_hidden = f*prev_hidden + i*x  (gates already relu'ed); emit f32 + bf16
__global__ void k_combine(const float* __restrict__ gates, const float* __restrict__ ph,
                          float* __restrict__ outHid, unsigned short* __restrict__ cur_bf) {
  size_t i = (size_t)blockIdx.x * blockDim.x + threadIdx.x;
  if (i < (size_t)B_ * H_) {
    int b = (int)(i >> 9);
    int h = (int)(i & (H_ - 1));
    size_t g = (size_t)b * 1536;
    float f  = gates[g + h];
    float it = gates[g + 512 + h];
    float x  = gates[g + 1024 + h];
    float c  = f * ph[i] + it * x;
    outHid[i] = c;
    cur_bf[i] = f2bf(c);
  }
}

// ---------------- WMMA GEMM ----------------
// C[M,N] = act( A0[M,K0] @ B0t[N,K0]^T + A1[M,K1] @ B1t[N,K1]^T + bias ) (* mul)
// A*, B*t are bf16 row-major. 128x256 block tile, 8 waves, each 64x64 (4x4 WMMA).
// Double-buffered LDS, async global->LDS staging.

#define BM 128
#define BN 256
#define BK 32
#define LSTR 48   // LDS row stride in bf16 elems: 96B -> every 32B fragment aligned

__global__ void __launch_bounds__(256)
k_wmma_gemm(const unsigned short* __restrict__ A0, const unsigned short* __restrict__ B0t, int K0,
            const unsigned short* __restrict__ A1, const unsigned short* __restrict__ B1t, int K1,
            const float* __restrict__ bias, const float* __restrict__ mulp,
            float* __restrict__ outF, unsigned short* __restrict__ outBF,
            int N, int doRelu) {
  __shared__ __align__(32) unsigned short As[2][BM * LSTR];
  __shared__ __align__(32) unsigned short Bs[2][BN * LSTR];

  const int tid  = threadIdx.x;
  const int lane = tid & 31;
  const int wave = tid >> 5;
  const int wr   = wave >> 2;    // 0..1 : 64-row slice
  const int wc   = wave & 3;     // 0..3 : 64-col slice
  const int hv   = lane >> 4;    // K-half / M-half selector
  const int l16  = lane & 15;

  const int m0 = blockIdx.y * BM;
  const int n0 = blockIdx.x * BN;

  const int steps0 = K0 / BK;
  const int nsteps = steps0 + K1 / BK;

  v8f acc[4][4];
#pragma unroll
  for (int mi = 0; mi < 4; ++mi)
#pragma unroll
    for (int ni = 0; ni < 4; ++ni)
      acc[mi][ni] = (v8f)0.0f;

  // Issue one k-step's tile staging (6 x B128 per thread: 2 for A, 4 for B).
  auto issue = [&](int s, int buf) {
    const unsigned short* A  = (s < steps0) ? A0 : A1;
    const unsigned short* Bt = (s < steps0) ? B0t : B1t;
    const int K  = (s < steps0) ? K0 : K1;
    const int kk = ((s < steps0) ? s : s - steps0) * BK;
#pragma unroll
    for (int i = 0; i < 2; ++i) {           // A tile: 128x32 bf16 = 512 uint4
      int idx = tid + i * 256;
      int row = idx >> 2;
      int c8  = (idx & 3) * 8;
      async_copy16(A + (size_t)(m0 + row) * K + kk + c8, &As[buf][row * LSTR + c8]);
    }
#pragma unroll
    for (int i = 0; i < 4; ++i) {           // B tile: 256x32 bf16 = 1024 uint4
      int idx = tid + i * 256;
      int row = idx >> 2;
      int c8  = (idx & 3) * 8;
      int rc  = n0 + row;
      if (rc > N - 1) rc = N - 1;           // clamp: garbage cols never stored
      async_copy16(Bt + (size_t)rc * K + kk + c8, &Bs[buf][row * LSTR + c8]);
    }
  };

  issue(0, 0);
  for (int s = 0; s < nsteps; ++s) {
    const int cur = s & 1;
    if (s + 1 < nsteps) {
      issue(s + 1, cur ^ 1);   // overlap next tile's loads with this step's WMMAs
      WAIT_ASYNC(6);           // oldest 6 (current buffer) complete
    } else {
      WAIT_ASYNC(0);
    }
    __syncthreads();

    v16bf bfrag[4];
#pragma unroll
    for (int ni = 0; ni < 4; ++ni) {
      int col = wc * 64 + ni * 16 + l16;
      bfrag[ni] = *(const v16bf*)&Bs[cur][col * LSTR + hv * 16];
    }
#pragma unroll
    for (int mi = 0; mi < 4; ++mi) {
      int row = wr * 64 + mi * 16 + l16;
      v16bf afrag = *(const v16bf*)&As[cur][row * LSTR + hv * 16];
#pragma unroll
      for (int ni = 0; ni < 4; ++ni) {
        acc[mi][ni] = __builtin_amdgcn_wmma_f32_16x16x32_bf16(
            false, afrag, false, bfrag[ni], (short)0, acc[mi][ni], false, false);
      }
    }
    __syncthreads();   // everyone done reading `cur` before it is re-staged
  }

  // Epilogue: C lane map: m = base + 8*hv + r, n = base + l16
#pragma unroll
  for (int ni = 0; ni < 4; ++ni) {
    int n = n0 + wc * 64 + ni * 16 + l16;
    if (n >= N) continue;
    float bv = bias ? bias[n] : 0.0f;
#pragma unroll
    for (int mi = 0; mi < 4; ++mi) {
      int mbase = m0 + wr * 64 + mi * 16 + hv * 8;
#pragma unroll
      for (int r = 0; r < 8; ++r) {
        float v = acc[mi][ni][r] + bv;
        if (doRelu) v = v > 0.0f ? v : 0.0f;
        size_t off = (size_t)(mbase + r) * N + n;
        if (mulp) v *= mulp[off];
        if (outF)  outF[off]  = v;
        if (outBF) outBF[off] = f2bf(v);
      }
    }
  }
}

// ---------------- launch ----------------

extern "C" void kernel_launch(void* const* d_in, const int* in_sizes, int n_in,
                              void* d_out, int out_size, void* d_ws, size_t ws_size,
                              hipStream_t stream) {
  (void)in_sizes; (void)n_in; (void)out_size; (void)ws_size;
  const int*   lab   = (const int*)  d_in[0];
  const float* ph    = (const float*)d_in[1];
  const float* img   = (const float*)d_in[2];
  const float* emb   = (const float*)d_in[3];
  const float* W_f_r = (const float*)d_in[4];
  const float* b_f_r = (const float*)d_in[5];
  const float* W_f_w = (const float*)d_in[6];
  const float* b_f_w = (const float*)d_in[7];
  const float* W_i_r = (const float*)d_in[8];
  const float* b_i_r = (const float*)d_in[9];
  const float* W_i_w = (const float*)d_in[10];
  const float* b_i_w = (const float*)d_in[11];
  const float* W_c_r = (const float*)d_in[12];
  const float* b_c_r = (const float*)d_in[13];
  const float* W_c_w = (const float*)d_in[14];
  const float* b_c_w = (const float*)d_in[15];
  const float* W_o_r = (const float*)d_in[16];
  const float* b_o_r = (const float*)d_in[17];
  const float* W_o_w = (const float*)d_in[18];
  const float* b_o_w = (const float*)d_in[19];
  const float* W_po  = (const float*)d_in[20];
  const float* b_po  = (const float*)d_in[21];
  const float* W_pi  = (const float*)d_in[22];
  const float* b_pi  = (const float*)d_in[23];

  float* out_score = (float*)d_out;                       // [B,V]
  float* out_hid   = out_score + (size_t)B_ * V_;         // [B,H]

  char* wp = (char*)d_ws;
  auto alloc = [&](size_t bytes) -> char* {
    char* p = wp;
    wp += (bytes + 255) & ~(size_t)255;
    return p;
  };
  unsigned short* ph_bf  = (unsigned short*)alloc((size_t)B_ * H_ * 2);
  unsigned short* wk_bf  = (unsigned short*)alloc((size_t)B_ * H_ * 2);
  unsigned short* img_bf = (unsigned short*)alloc((size_t)B_ * F_ * 2);
  unsigned short* emb_bf = (unsigned short*)alloc((size_t)V_ * H_ * 2);
  unsigned short* cur_bf = (unsigned short*)alloc((size_t)B_ * H_ * 2);
  unsigned short* o_bf   = (unsigned short*)alloc((size_t)B_ * H_ * 2);
  unsigned short* xp_bf  = (unsigned short*)alloc((size_t)B_ * H_ * 2);
  unsigned short* WtGr   = (unsigned short*)alloc((size_t)3 * H_ * H_ * 2); // [1536,512]
  unsigned short* WtGw   = (unsigned short*)alloc((size_t)3 * H_ * H_ * 2);
  unsigned short* WtOr   = (unsigned short*)alloc((size_t)H_ * H_ * 2);
  unsigned short* WtOw   = (unsigned short*)alloc((size_t)H_ * H_ * 2);
  unsigned short* WtPo   = (unsigned short*)alloc((size_t)H_ * H_ * 2);
  unsigned short* WtPi   = (unsigned short*)alloc((size_t)H_ * F_ * 2);     // [512,2048]
  float*          biasG  = (float*)alloc((size_t)3 * H_ * 4);
  float*          biasO  = (float*)alloc((size_t)H_ * 4);
  float*          biasP  = (float*)alloc((size_t)H_ * 4);
  float*          gates  = (float*)alloc((size_t)B_ * 3 * H_ * 4);          // [B,1536]

  const int TB = 256;
  auto blocks = [](size_t n) { return (unsigned)((n + 255) / 256); };

  // --- prep ---
  k_f2bf<<<blocks((size_t)B_ * H_), TB, 0, stream>>>(ph,  ph_bf,  (size_t)B_ * H_);
  k_f2bf<<<blocks((size_t)B_ * F_), TB, 0, stream>>>(img, img_bf, (size_t)B_ * F_);
  k_f2bf<<<blocks((size_t)V_ * H_), TB, 0, stream>>>(emb, emb_bf, (size_t)V_ * H_);
  k_gather_bf<<<blocks((size_t)B_ * H_), TB, 0, stream>>>(lab, emb, wk_bf);

  unsigned tw = blocks((size_t)H_ * H_);
  k_transpose_bf<<<tw, TB, 0, stream>>>(W_f_r, WtGr,                   H_, H_);
  k_transpose_bf<<<tw, TB, 0, stream>>>(W_i_r, WtGr + (size_t)H_*H_,   H_, H_);
  k_transpose_bf<<<tw, TB, 0, stream>>>(W_c_r, WtGr + (size_t)2*H_*H_, H_, H_);
  k_transpose_bf<<<tw, TB, 0, stream>>>(W_f_w, WtGw,                   H_, H_);
  k_transpose_bf<<<tw, TB, 0, stream>>>(W_i_w, WtGw + (size_t)H_*H_,   H_, H_);
  k_transpose_bf<<<tw, TB, 0, stream>>>(W_c_w, WtGw + (size_t)2*H_*H_, H_, H_);
  k_transpose_bf<<<tw, TB, 0, stream>>>(W_o_r, WtOr, H_, H_);
  k_transpose_bf<<<tw, TB, 0, stream>>>(W_o_w, WtOw, H_, H_);
  k_transpose_bf<<<tw, TB, 0, stream>>>(W_po,  WtPo, H_, H_);
  k_transpose_bf<<<blocks((size_t)F_ * H_), TB, 0, stream>>>(W_pi, WtPi, F_, H_);

  k_vadd<<<2, TB, 0, stream>>>(b_f_r, b_f_w, biasG,          H_);
  k_vadd<<<2, TB, 0, stream>>>(b_i_r, b_i_w, biasG + H_,     H_);
  k_vadd<<<2, TB, 0, stream>>>(b_c_r, b_c_w, biasG + 2 * H_, H_);
  k_vadd<<<2, TB, 0, stream>>>(b_o_r, b_o_w, biasO, H_);
  k_vadd<<<2, TB, 0, stream>>>(b_po,  b_pi,  biasP, H_);

  const dim3 blk(256);
  // GEMM1: gates[B,1536] = relu(ph@W_gr + wk@W_gw + biasG)
  k_wmma_gemm<<<dim3(6, B_ / BM), blk, 0, stream>>>(
      ph_bf, WtGr, H_, wk_bf, WtGw, H_, biasG, nullptr, gates, nullptr, 3 * H_, 1);
  // cur_hidden = f*ph + i*x  -> out_hid (f32) + cur_bf
  k_combine<<<blocks((size_t)B_ * H_), TB, 0, stream>>>(gates, ph, out_hid, cur_bf);
  // GEMM2: o_t = cur_hidden * relu(cur@W_or + wk@W_ow + biasO) -> o_bf
  k_wmma_gemm<<<dim3(2, B_ / BM), blk, 0, stream>>>(
      cur_bf, WtOr, H_, wk_bf, WtOw, H_, biasO, out_hid, nullptr, o_bf, H_, 1);
  // GEMM3: x_proj = relu(o_t@W_po + img@W_pi + biasP) -> xp_bf
  k_wmma_gemm<<<dim3(2, B_ / BM), blk, 0, stream>>>(
      o_bf, WtPo, H_, img_bf, WtPi, F_, biasP, nullptr, nullptr, xp_bf, H_, 1);
  // GEMM4: score[B,1000] = x_proj @ emb^T  (emb rows are B-matrix columns directly)
  k_wmma_gemm<<<dim3(4, B_ / BM), blk, 0, stream>>>(
      xp_bf, emb_bf, H_, nullptr, nullptr, 0, nullptr, nullptr, out_score, nullptr, V_, 0);
}
